// GroupedQueryAttention_18253611008341
// MI455X (gfx1250) — matrix operand order repaired and checked by reading, not compiled, not run on hardware
//
#include <hip/hip_runtime.h>
#include <hip/hip_bf16.h>
#include <stdint.h>

// Problem constants (reference: B=4, S=2048, DIM=2048, 16 heads x 128)
#define DIM     2048
#define HD      128
#define TOKENS  8192   // B*S

typedef __attribute__((ext_vector_type(16))) __bf16 v16bf;
typedef __attribute__((ext_vector_type(8)))  float  v8f;

union FragAB { v16bf v; uint4 q[2]; };

static constexpr int LDT = 40;   // padded LDS row stride in halves (80B: conflict-free b128)

__device__ __forceinline__ unsigned short f32_to_bf16(float f) {
  unsigned u = __float_as_uint(f);
  unsigned r = u + 0x7FFFu + ((u >> 16) & 1u);   // round-to-nearest-even
  return (unsigned short)(r >> 16);
}

__device__ __forceinline__ v8f v8f_zero() {
  v8f z;
#pragma unroll
  for (int i = 0; i < 8; ++i) z[i] = 0.0f;
  return z;
}

// ---- CDNA5 async global->LDS copy (ASYNCcnt path), 16B per lane ----
__device__ __forceinline__ void async_b128_to_lds(unsigned lds_off,
                                                  const unsigned short* gptr) {
  asm volatile("global_load_async_to_lds_b128 %0, %1, off"
               :: "v"(lds_off), "v"((unsigned long long)(uintptr_t)gptr)
               : "memory");
}
__device__ __forceinline__ void wait_asynccnt0() {
  asm volatile("s_wait_asynccnt 0x0" ::: "memory");
}

// ---- wave32 16-lane XOR reductions via ds_swizzle (and_mask=0x1f) ----
template <int XORM>
__device__ __forceinline__ float swz_xor(float v) {
  return __int_as_float(
      __builtin_amdgcn_ds_swizzle(__float_as_int(v), 0x7C00 | XORM));
}
__device__ __forceinline__ float red_max16(float v) {
  v = fmaxf(v, swz_xor<8>(v));
  v = fmaxf(v, swz_xor<4>(v));
  v = fmaxf(v, swz_xor<2>(v));
  v = fmaxf(v, swz_xor<1>(v));
  return v;
}
__device__ __forceinline__ float red_sum16(float v) {
  v += swz_xor<8>(v);
  v += swz_xor<4>(v);
  v += swz_xor<2>(v);
  v += swz_xor<1>(v);
  return v;
}

// ---------------------------------------------------------------------------
// fp32 -> bf16 conversion pass (one-shot; operands then live bf16 in L2)
// ---------------------------------------------------------------------------
__global__ __launch_bounds__(256)
void cvt_f32_bf16(const float* __restrict__ in, unsigned short* __restrict__ out,
                  int n4) {
  int i = blockIdx.x * 256 + threadIdx.x;
  int stride = gridDim.x * 256;
  for (; i < n4; i += stride) {
    float4 d = ((const float4*)in)[i];
    uint2 pk = make_uint2(((unsigned)f32_to_bf16(d.y) << 16) | f32_to_bf16(d.x),
                          ((unsigned)f32_to_bf16(d.w) << 16) | f32_to_bf16(d.z));
    ((uint2*)out)[i] = pk;
  }
}

// ---------------------------------------------------------------------------
// Tiled WMMA GEMM:  Out[M,N] = A[M,K] @ W[N,K]^T + bias[N]   (A, W in bf16)
//   Block: 256 threads (8 waves), tile 128x128, K-step 32.
//   Double-buffered LDS filled by async global->LDS b128 copies; the next
//   tile's fill overlaps the current tile's 8 WMMAs per wave.
// ---------------------------------------------------------------------------
template <bool OUT_BF16>
__global__ __launch_bounds__(256)
void gemm_wmma(const unsigned short* __restrict__ A,
               const unsigned short* __restrict__ W,
               const float* __restrict__ bias, void* __restrict__ Out,
               int M, int N, int K)
{
  __shared__ unsigned short As[2][128 * LDT];
  __shared__ unsigned short Bs[2][128 * LDT];

  const int tid    = threadIdx.x;
  const int lane   = tid & 31;
  const int wave   = tid >> 5;
  const int wm     = wave >> 2;                 // 0..1
  const int wn     = wave & 3;                  // 0..3
  const int half16 = lane >> 4;                 // 0/1 (16-lane half)
  const int l16    = lane & 15;
  const int m0     = blockIdx.y * 128;
  const int n0     = blockIdx.x * 128;

  // this thread's two b128 groups per 128x32 tile
  const int g0   = tid;                         // group: row=g>>2, colh=(g&3)*8
  const int row0 = g0 >> 2,         colh0 = (g0 & 3) << 3;
  const int row1 = (g0 + 256) >> 2, colh1 = ((g0 + 256) & 3) << 3;

  const unsigned short* Abase = A + (size_t)m0 * K;
  const unsigned short* Wbase = W + (size_t)n0 * K;

  auto stage = [&](int buf, int k0) {
    unsigned as = (unsigned)(uintptr_t)&As[buf][0];
    unsigned bs = (unsigned)(uintptr_t)&Bs[buf][0];
    async_b128_to_lds(as + (unsigned)(row0 * LDT + colh0) * 2,
                      Abase + (size_t)row0 * K + k0 + colh0);
    async_b128_to_lds(as + (unsigned)(row1 * LDT + colh1) * 2,
                      Abase + (size_t)row1 * K + k0 + colh1);
    async_b128_to_lds(bs + (unsigned)(row0 * LDT + colh0) * 2,
                      Wbase + (size_t)row0 * K + k0 + colh0);
    async_b128_to_lds(bs + (unsigned)(row1 * LDT + colh1) * 2,
                      Wbase + (size_t)row1 * K + k0 + colh1);
  };

  v8f acc[4][2];
#pragma unroll
  for (int i = 0; i < 4; ++i)
#pragma unroll
    for (int j = 0; j < 2; ++j) acc[i][j] = v8f_zero();

  stage(0, 0);   // prime the pipeline

  for (int k0 = 0; k0 < K; k0 += 32) {
    const int cur = (k0 >> 5) & 1;
    wait_asynccnt0();        // this wave's async fills complete
    __syncthreads();         // all waves' fills of buf[cur] visible
    if (k0 + 32 < K) stage(cur ^ 1, k0 + 32);   // overlap next fill with WMMA

    // A fragment (16-bit 16x32): row=l16; e<8 -> K=e+8*half16; e>=8 -> +16
    FragAB a[4];
#pragma unroll
    for (int fm = 0; fm < 4; ++fm) {
      const unsigned short* p = &As[cur][(wm * 64 + fm * 16 + l16) * LDT + half16 * 8];
      a[fm].q[0] = *(const uint4*)(p);
      a[fm].q[1] = *(const uint4*)(p + 16);
    }
    // B fragment (16-bit 32x16): col=l16; e<8 -> K=e+8*half16; e>=8 -> +16
    FragAB b[2];
#pragma unroll
    for (int fn = 0; fn < 2; ++fn) {
      const unsigned short* p = &Bs[cur][(wn * 32 + fn * 16 + l16) * LDT + half16 * 8];
      b[fn].q[0] = *(const uint4*)(p);
      b[fn].q[1] = *(const uint4*)(p + 16);
    }

#pragma unroll
    for (int fm = 0; fm < 4; ++fm)
#pragma unroll
      for (int fn = 0; fn < 2; ++fn)
        acc[fm][fn] = __builtin_amdgcn_wmma_f32_16x16x32_bf16(
            false, a[fm].v, false, b[fn].v, (short)0, acc[fm][fn], false, false);

    __syncthreads();         // everyone done reading buf[cur] before refill
  }

  // ---- epilogue: bias + store ----
#pragma unroll
  for (int fn = 0; fn < 2; ++fn) {
    int n = n0 + wn * 32 + fn * 16 + l16;
    float bn = bias[n];
#pragma unroll
    for (int fm = 0; fm < 4; ++fm) {
#pragma unroll
      for (int r = 0; r < 8; ++r) {
        int m = m0 + wm * 64 + fm * 16 + r + half16 * 8;   // C layout
        float val = acc[fm][fn][r] + bn;
        if (OUT_BF16)
          ((unsigned short*)Out)[(size_t)m * N + n] = f32_to_bf16(val);
        else
          ((float*)Out)[(size_t)m * N + n] = val;
      }
    }
  }
}

// ---------------------------------------------------------------------------
// Per-token head-vs-head attention. One wave per token (8 waves / block).
//   scores[16,16] = q(16x128) @ k(16x128)^T / sqrt(128)  -> 4 WMMAs
//   softmax over the k-head axis (ds_swizzle XOR reductions)
//   ctx(16x128)   = attn(16x16, K zero-padded to 32) @ v -> 8 WMMAs
// Ctx may alias Q: each wave fully consumes its token's q before writing ctx,
// and tokens map 1:1 to waves.
// ---------------------------------------------------------------------------
__global__ __launch_bounds__(256)
void attn_headmix(const unsigned short* __restrict__ Q,
                  const unsigned short* __restrict__ Km,
                  const unsigned short* __restrict__ V,
                  unsigned short* __restrict__ Ctx)
{
  __shared__ unsigned short vT[8][HD * 16];     // per-wave V^T  [d][h]
  __shared__ unsigned short attnS[8][16 * 16];  // per-wave attn [m][n] bf16

  const int tid    = threadIdx.x;
  const int lane   = tid & 31;
  const int wave   = tid >> 5;
  const int half16 = lane >> 4;
  const int l16    = lane & 15;
  const int t      = blockIdx.x * 8 + wave;

  const unsigned short* q = Q  + (size_t)t * DIM;
  const unsigned short* k = Km + (size_t)t * DIM;
  const unsigned short* v = V  + (size_t)t * DIM;

  // stage V transposed: vT[d][h] = v[h*128 + d]  (coalesced global reads)
  for (int i = 0; i < 64; ++i) {
    int e = i * 32 + lane;                      // 0..2047
    vT[wave][(e & 127) * 16 + (e >> 7)] = v[e];
  }

  // ---- scores = q @ k^T ----
  v8f sc = v8f_zero();
#pragma unroll
  for (int kk = 0; kk < 4; ++kk) {
    FragAB a, b;
    const unsigned short* qa = q + l16 * HD + kk * 32 + half16 * 8;
    a.q[0] = *(const uint4*)(qa);
    a.q[1] = *(const uint4*)(qa + 16);
    const unsigned short* kb = k + l16 * HD + kk * 32 + half16 * 8;
    b.q[0] = *(const uint4*)(kb);
    b.q[1] = *(const uint4*)(kb + 16);
    sc = __builtin_amdgcn_wmma_f32_16x16x32_bf16(
        false, a.v, false, b.v, (short)0, sc, false, false);
  }

  // ---- row softmax: rows live across 16 lanes (XOR masks stay in-half) ----
  const float scale = 0.08838834764831845f;     // 1/sqrt(128)
#pragma unroll
  for (int r = 0; r < 8; ++r) {
    float s  = sc[r] * scale;
    float mx = red_max16(s);
    float e  = __expf(s - mx);
    float p  = e / red_sum16(e);
    attnS[wave][(r + half16 * 8) * 16 + l16] = f32_to_bf16(p);   // [m][n]
  }

  __syncthreads();   // uniform; publishes vT + attnS for fragment-layout reads

  // ---- context = attn_pad(16x32) @ [v ; 0](32x128) ----
  FragAB a;
  {
    const unsigned short* pa = &attnS[wave][l16 * 16 + half16 * 8];
    a.q[0] = *(const uint4*)(pa);               // K<16: real attn
    a.q[1] = make_uint4(0, 0, 0, 0);            // K>=16: zero pad
  }
  const uint4 z4 = make_uint4(0, 0, 0, 0);
#pragma unroll
  for (int dt = 0; dt < 8; ++dt) {
    FragAB b;
    const unsigned short* pb = &vT[wave][(dt * 16 + l16) * 16];
    uint4 b0 = *(const uint4*)(pb);
    uint4 b1 = *(const uint4*)(pb + 8);
    b.q[0] = half16 ? b1 : b0;                  // lanes 16-31 hold K=8..15 in e<8
    b.q[1] = z4;                                // e>=8 is K>=16 -> 0; EXEC stays full
    v8f c = v8f_zero();
    c = __builtin_amdgcn_wmma_f32_16x16x32_bf16(
        false, a.v, false, b.v, (short)0, c, false, false);
#pragma unroll
    for (int r = 0; r < 8; ++r)
      Ctx[(size_t)t * DIM + (r + half16 * 8) * HD + dt * 16 + l16] =
          f32_to_bf16(c[r]);
  }
}

// ---------------------------------------------------------------------------
extern "C" void kernel_launch(void* const* d_in, const int* in_sizes, int n_in,
                              void* d_out, int out_size, void* d_ws, size_t ws_size,
                              hipStream_t stream)
{
  const float* x  = (const float*)d_in[0];
  const float* Wq = (const float*)d_in[1];
  const float* bq = (const float*)d_in[2];
  const float* Wk = (const float*)d_in[3];
  const float* bk = (const float*)d_in[4];
  const float* Wv = (const float*)d_in[5];
  const float* bv = (const float*)d_in[6];
  const float* Wo = (const float*)d_in[7];
  const float* bo = (const float*)d_in[8];

  // workspace layout (bf16 halves), 160 MB total:
  //   [x 32MB][Wq 8MB][Wk 8MB][Wv 8MB][Wo 8MB][q 32MB][k 32MB][v 32MB]
  // ctx aliases the q plane (safe: per-wave read-q-then-write-ctx, 1 wave/token)
  const size_t XE = (size_t)TOKENS * DIM;   // elements in an activation plane
  const size_t WE = (size_t)DIM * DIM;      // elements in a weight plane
  unsigned short* xbf = (unsigned short*)d_ws;
  unsigned short* wqb = xbf + XE;
  unsigned short* wkb = wqb + WE;
  unsigned short* wvb = wkb + WE;
  unsigned short* wob = wvb + WE;
  unsigned short* qb  = wob + WE;
  unsigned short* kb  = qb + XE;
  unsigned short* vb  = kb + XE;
  unsigned short* cb  = qb;                 // alias

  dim3 blk(256);

  // one-shot fp32 -> bf16 conversion of all GEMM operands
  cvt_f32_bf16<<<dim3(4096), blk, 0, stream>>>(x,  xbf, (int)(XE / 4));
  cvt_f32_bf16<<<dim3(1024), blk, 0, stream>>>(Wq, wqb, (int)(WE / 4));
  cvt_f32_bf16<<<dim3(1024), blk, 0, stream>>>(Wk, wkb, (int)(WE / 4));
  cvt_f32_bf16<<<dim3(1024), blk, 0, stream>>>(Wv, wvb, (int)(WE / 4));
  cvt_f32_bf16<<<dim3(1024), blk, 0, stream>>>(Wo, wob, (int)(WE / 4));

  dim3 grid(DIM / 128, TOKENS / 128);
  gemm_wmma<true><<<grid, blk, 0, stream>>>(xbf, wqb, bq, qb, TOKENS, DIM, DIM);
  gemm_wmma<true><<<grid, blk, 0, stream>>>(xbf, wkb, bk, kb, TOKENS, DIM, DIM);
  gemm_wmma<true><<<grid, blk, 0, stream>>>(xbf, wvb, bv, vb, TOKENS, DIM, DIM);

  attn_headmix<<<dim3(TOKENS / 8), blk, 0, stream>>>(qb, kb, vb, cb);

  gemm_wmma<false><<<grid, blk, 0, stream>>>(cb, wob, bo, (float*)d_out,
                                             TOKENS, DIM, DIM);
}
